// VectorQuantizerG_84980222919423
// MI455X (gfx1250) — matrix operand order repaired and checked by reading, not compiled, not run on hardware
//
#include <hip/hip_runtime.h>
#include <hip/hip_bf16.h>

// Problem constants (from reference): N=65536, G=4, K=1024, C=64
#define VQ_N 65536
#define VQ_G 4
#define VQ_K 1024
#define VQ_C 64
#define ROW_STRIDE (VQ_G * VQ_C)   // 256 floats per z row
#define BLOCKS_X 512               // N / 128 rows per block
#define NPARTIAL (BLOCKS_X * VQ_G) // 2048 block partial sums

typedef _Float16 v16h __attribute__((ext_vector_type(16)));
typedef _Float16 v8h  __attribute__((ext_vector_type(8)));
typedef _Float16 h4   __attribute__((ext_vector_type(4)));
typedef float    v8f  __attribute__((ext_vector_type(8)));
typedef float    f8v  __attribute__((ext_vector_type(8)));

// ds_swizzle_b32 xor-butterfly within 32 lanes: lane reads lane^M.
// offset encoding (group-of-32): [14:10]=xor_mask, [9:5]=or_mask, [4:0]=and_mask
template <int M>
__device__ __forceinline__ float swz_xor(float x) {
    int r = __builtin_amdgcn_ds_swizzle(__builtin_bit_cast(int, x), (M << 10) | 0x1f);
    return __builtin_bit_cast(float, r);
}

// ---------------------------------------------------------------------------
// Kernel 1: convert emb (G*K*C fp32) -> f16 copy in ws, and e2[g*K+k] = ||e||^2
// 65536 threads, 4 elements each; 16 threads per codebook row.
// ---------------------------------------------------------------------------
__global__ __launch_bounds__(256) void vq_prep(const float* __restrict__ emb,
                                               _Float16* __restrict__ embh,
                                               float* __restrict__ e2) {
    int t = blockIdx.x * 256 + threadIdx.x;          // 0 .. 65535
    float4 v = ((const float4*)emb)[t];
    h4 h;
    h.x = (_Float16)v.x; h.y = (_Float16)v.y; h.z = (_Float16)v.z; h.w = (_Float16)v.w;
    ((h4*)embh)[t] = h;
    float ss = v.x * v.x + v.y * v.y + v.z * v.z + v.w * v.w;
    // reduce across the 16 lanes holding one row (64/4 = 16 threads per row)
    ss += swz_xor<1>(ss);
    ss += swz_xor<2>(ss);
    ss += swz_xor<4>(ss);
    ss += swz_xor<8>(ss);
    if ((t & 15) == 0)
        e2[t >> 4] = ss;
}

// ---------------------------------------------------------------------------
// Kernel 2: WMMA distance + packed argmin + gather + STE output + loss partial.
// grid = (512, G); block = 256 (8 waves); each wave: 16 rows x all K, one g.
// Packed argmin: low 10 mantissa bits of d replaced by k (exact, K=1024=2^10);
// mangling error <= 1024 ulps of |d|~1e-2 ~ 1e-6, below the f16 noise floor.
// ---------------------------------------------------------------------------
__global__ __launch_bounds__(256) void vq_main(const float* __restrict__ z,
                                               const float* __restrict__ embf,
                                               const _Float16* __restrict__ embh,
                                               const float* __restrict__ e2,
                                               float* __restrict__ out,
                                               float* __restrict__ partial) {
    const int g    = blockIdx.y;
    const int wave = threadIdx.x >> 5;
    const int lane = threadIdx.x & 31;
    const int lo   = lane & 15;
    const int hi   = lane >> 4;
    const int row0 = blockIdx.x * 128 + wave * 16;   // first n of this wave's tile

    // ---- A fragments: 16x32, this code's contiguous k order (same as B).
    // halfs 0-15 : K = 16*hi + h -> floats  c0 + 16*hi .. +15
    const float* zrow = z + (size_t)(row0 + lo) * ROW_STRIDE + g * VQ_C;
    f8v f00 = *(const f8v*)(zrow +      16 * hi);
    f8v f01 = *(const f8v*)(zrow +  8 + 16 * hi);
    f8v f10 = *(const f8v*)(zrow + 32 + 16 * hi);
    f8v f11 = *(const f8v*)(zrow + 40 + 16 * hi);
    v8h h00 = __builtin_convertvector(f00, v8h);
    v8h h01 = __builtin_convertvector(f01, v8h);
    v8h h10 = __builtin_convertvector(f10, v8h);
    v8h h11 = __builtin_convertvector(f11, v8h);
    v16h a0 = __builtin_shufflevector(h00, h01, 0,1,2,3,4,5,6,7,8,9,10,11,12,13,14,15);
    v16h a1 = __builtin_shufflevector(h10, h11, 0,1,2,3,4,5,6,7,8,9,10,11,12,13,14,15);

    // packed (distance | index) running minima, one per row held by this lane
    float best[8];
    #pragma unroll
    for (int v = 0; v < 8; ++v)
        best[v] = __builtin_bit_cast(float, 0x7F7FFC00u);  // FLT_MAX, low 10 bits clear

    const _Float16* ebase = embh + (size_t)g * VQ_K * VQ_C;
    const float*    e2g   = e2 + g * VQ_K;

    // 64 k-chunks of 16 codewords; unroll by 2 with independent acc chains so
    // the scheduler hides WMMA->VALU hazards with the sibling chunk's WMMAs.
    for (int kc = 0; kc < VQ_K / 16; kc += 2) {
        const int kA = kc * 16 + lo;
        const int kB = kA + 16;
        const _Float16* erowA = ebase + (size_t)kA * VQ_C;
        const _Float16* erowB = ebase + (size_t)kB * VQ_C;
        // B fragments: 32x16 in this code's contiguous k order (as A): halfs 0..15 <-> K = 16*hi + h
        v16h bA0 = *(const v16h*)(erowA +      16 * hi);
        v16h bA1 = *(const v16h*)(erowA + 32 + 16 * hi);
        v16h bB0 = *(const v16h*)(erowB +      16 * hi);
        v16h bB1 = *(const v16h*)(erowB + 32 + 16 * hi);
        float e2A = e2g[kA];
        float e2B = e2g[kB];

        v8f accA = {};
        v8f accB = {};
        accA = __builtin_amdgcn_wmma_f32_16x16x32_f16(false, a0, false, bA0,
                                                      (short)0, accA, false, false);
        accB = __builtin_amdgcn_wmma_f32_16x16x32_f16(false, a0, false, bB0,
                                                      (short)0, accB, false, false);
        accA = __builtin_amdgcn_wmma_f32_16x16x32_f16(false, a1, false, bA1,
                                                      (short)0, accA, false, false);
        accB = __builtin_amdgcn_wmma_f32_16x16x32_f16(false, a1, false, bB1,
                                                      (short)0, accB, false, false);

        // D layout: VGPR v -> row M = 8*hi + v, col N = lo
        #pragma unroll
        for (int v = 0; v < 8; ++v) {
            float dA = fmaf(-2.0f, accA[v], e2A);
            float dB = fmaf(-2.0f, accB[v], e2B);
            // v_bfi_b32: (d & ~0x3FF) | (k & 0x3FF)
            uint32_t pA = (__builtin_bit_cast(uint32_t, dA) & 0xFFFFFC00u) |
                          ((uint32_t)kA & 0x3FFu);
            uint32_t pB = (__builtin_bit_cast(uint32_t, dB) & 0xFFFFFC00u) |
                          ((uint32_t)kB & 0x3FFu);
            // v_min3_f32
            best[v] = fminf(best[v], fminf(__builtin_bit_cast(float, pA),
                                           __builtin_bit_cast(float, pB)));
        }
    }

    // ---- butterfly min across the 16 lanes of each half-wave (ds_swizzle)
    #pragma unroll
    for (int v = 0; v < 8; ++v) {
        best[v] = fminf(best[v], swz_xor<1>(best[v]));
        best[v] = fminf(best[v], swz_xor<2>(best[v]));
        best[v] = fminf(best[v], swz_xor<4>(best[v]));
        best[v] = fminf(best[v], swz_xor<8>(best[v]));
    }

    // ---- gather fp32 codeword, STE output, loss partial
    float lsum = 0.0f;
    #pragma unroll
    for (int v = 0; v < 8; ++v) {
        const int rowm = hi * 8 + v;
        const int idx  = (int)(__builtin_bit_cast(uint32_t, best[v]) & 0x3FFu);
        const float* er   = embf + ((size_t)(g * VQ_K + idx)) * VQ_C;
        const float* zr   = z    + (size_t)(row0 + rowm) * ROW_STRIDE + g * VQ_C;
        float*       orow = out  + (size_t)(row0 + rowm) * ROW_STRIDE + g * VQ_C;
        #pragma unroll
        for (int t = 0; t < 4; ++t) {
            const int c = lo + 16 * t;
            float e  = er[c];
            float zv = zr[c];
            orow[c] = zv + (e - zv);                 // matches reference STE arithmetic
            float dd = e - zv;
            lsum += dd * dd;
        }
    }

    // ---- deterministic block reduction of loss partials
    __shared__ float red[256];
    red[threadIdx.x] = lsum;
    __syncthreads();
    #pragma unroll
    for (int s = 128; s > 0; s >>= 1) {
        if (threadIdx.x < s) red[threadIdx.x] += red[threadIdx.x + s];
        __syncthreads();
    }
    if (threadIdx.x == 0)
        partial[blockIdx.y * BLOCKS_X + blockIdx.x] = red[0];
}

// ---------------------------------------------------------------------------
// Kernel 3: fixed-order reduction of 2048 partials -> loss scalar
// ---------------------------------------------------------------------------
__global__ __launch_bounds__(256) void vq_loss(const float* __restrict__ partial,
                                               float* __restrict__ loss_out) {
    __shared__ float red[256];
    float s = 0.0f;
    for (int i = threadIdx.x; i < NPARTIAL; i += 256) s += partial[i];
    red[threadIdx.x] = s;
    __syncthreads();
    #pragma unroll
    for (int st = 128; st > 0; st >>= 1) {
        if (threadIdx.x < st) red[threadIdx.x] += red[threadIdx.x + st];
        __syncthreads();
    }
    if (threadIdx.x == 0)
        *loss_out = red[0] * (1.5f / (float)((size_t)VQ_N * VQ_G * VQ_C));
}

extern "C" void kernel_launch(void* const* d_in, const int* in_sizes, int n_in,
                              void* d_out, int out_size, void* d_ws, size_t ws_size,
                              hipStream_t stream) {
    const float* z   = (const float*)d_in[0];   // (N, G*C) fp32
    const float* emb = (const float*)d_in[1];   // (G, K, C) fp32
    float* out = (float*)d_out;                 // zq_ste flat + loss scalar

    // workspace layout
    _Float16* embh    = (_Float16*)d_ws;                              // 512 KB
    float*    e2      = (float*)((char*)d_ws + 524288);               //  16 KB
    float*    partial = (float*)((char*)d_ws + 524288 + 16384);       //   8 KB

    vq_prep<<<dim3(256), dim3(256), 0, stream>>>(emb, embh, e2);
    vq_main<<<dim3(BLOCKS_X, VQ_G), dim3(256), 0, stream>>>(z, emb, embh, e2, out, partial);
    vq_loss<<<dim3(1), dim3(256), 0, stream>>>(partial, out + (out_size - 1));
}
